// GSCL_14748917694891
// MI455X (gfx1250) — compile-verified, matrix operand order checked by reading, not run on hardware
//
#include <hip/hip_runtime.h>
#include <hip/hip_bf16.h>
#include <math.h>

typedef __attribute__((ext_vector_type(16))) _Float16 v16h;
typedef __attribute__((ext_vector_type(8)))  _Float16 v8h;
typedef __attribute__((ext_vector_type(8)))  float    v8f;

#define NN   8192
#define EE   262144
#define FDIM 128        // projection/act dim
#define TEMP_INV 2.0f   // 1/0.5
#define NCT  (NN / 16)  // 512 column tiles
#define BROW 136        // 128 halves + 8 pad -> 272B row stride, conflict-free

// ---------------------------------------------------------------------------
// Generic fused GEMM: C[M,Nc] = act(A[M,K] @ B[K,Nc] + bias)
// f32 operands converted to f16 in registers, v_wmma_f32_16x16x32_f16 core.
// One wave computes one 16x16 C tile. K % 32 == 0, Nc % 16 == 0, M % 16 == 0.
// act: 0 = none, 1 = relu, 2 = elu
// ---------------------------------------------------------------------------
__global__ void gemm_wmma_f16(const float* __restrict__ A,
                              const float* __restrict__ B,
                              const float* __restrict__ bias,
                              float* __restrict__ C,
                              int M, int K, int Nc, int act) {
  const int lane = threadIdx.x & 31;
  const int wave = threadIdx.x >> 5;
  const int ntn  = Nc >> 4;
  const int tot  = (M >> 4) * ntn;
  const int tile = blockIdx.x * (blockDim.x >> 5) + wave;
  if (tile >= tot) return;
  const int tm = tile / ntn;
  const int tn = tile - tm * ntn;
  const int g = lane >> 4;      // lane group (0: lanes 0-15, 1: lanes 16-31)
  const int r = lane & 15;

  const float* arow = A + (size_t)(tm * 16 + r) * K;   // A row for this lane
  const int    cb   = tn * 16 + r;                     // B/C column for this lane

  v8f c = {};
  for (int kb = 0; kb < K; kb += 32) {
    v16h a, b;
    // A layout: half h -> k = kb + 16*(h/8) + 8*g + (h%8); contiguous runs of 8
    const float* pa0 = arow + kb + 8 * g;
    const float* pa1 = pa0 + 16;
#pragma unroll
    for (int i = 0; i < 8; ++i) {
      a[i]     = (_Float16)pa0[i];
      a[8 + i] = (_Float16)pa1[i];
    }
    // B layout mirrors A with n = lane%16: b[h] = B[k(h)][cb]
#pragma unroll
    for (int h = 0; h < 16; ++h) {
      const int k = kb + ((h < 8) ? (8 * g + h) : (16 + 8 * g + (h - 8)));
      b[h] = (_Float16)B[(size_t)k * Nc + cb];
    }
    c = __builtin_amdgcn_wmma_f32_16x16x32_f16(false, a, false, b,
                                               (short)0, c, false, false);
  }

  const float bv = bias ? bias[cb] : 0.0f;
#pragma unroll
  for (int v = 0; v < 8; ++v) {
    float val = c[v] + bv;
    if (act == 1)      val = fmaxf(val, 0.0f);
    else if (act == 2) val = (val > 0.0f) ? val : (__expf(val) - 1.0f);
    const int m = tm * 16 + v + 8 * g;   // C row mapping per ISA layout
    C[(size_t)m * Nc + cb] = val;
  }
}

// ---------------------------------------------------------------------------
// out[i*F + f] = bias[f]   (F is a power of two)
// ---------------------------------------------------------------------------
__global__ void bias_init(float* __restrict__ out, const float* __restrict__ bias,
                          int fmask, int total) {
  int i = blockIdx.x * blockDim.x + threadIdx.x;
  if (i < total) out[i] = bias[i & fmask];
}

// ---------------------------------------------------------------------------
// GCN scatter: out[row[e]] += w[e] * sup[col[e]]  (atomic f32)
// ---------------------------------------------------------------------------
__global__ void gcn_scatter(const int* __restrict__ ei, const float* __restrict__ ew,
                            const float* __restrict__ sup, float* __restrict__ out,
                            int E, int F) {
  int e = blockIdx.x * blockDim.x + threadIdx.x;
  if (e >= E) return;
  const int row = ei[e];
  const int col = ei[E + e];
  const float w = ew[e];
  const float* s = sup + (size_t)col * F;
  float* o = out + (size_t)row * F;
  for (int f = 0; f < F; ++f) atomicAdd(o + f, w * s[f]);
}

__global__ void relu_inplace(float* __restrict__ x, int total) {
  int i = blockIdx.x * blockDim.x + threadIdx.x;
  if (i < total) x[i] = fmaxf(x[i], 0.0f);
}

// ---------------------------------------------------------------------------
// Row L2 normalize (F = 128, one block per row), write f32 in place + f16 copy
// ---------------------------------------------------------------------------
__global__ void normalize_rows(float* __restrict__ z, _Float16* __restrict__ zh) {
  __shared__ float red[FDIM];
  const int i = blockIdx.x, t = threadIdx.x;
  float v = z[(size_t)i * FDIM + t];
  red[t] = v * v;
  __syncthreads();
  for (int s = FDIM / 2; s > 0; s >>= 1) {
    if (t < s) red[t] += red[t + s];
    __syncthreads();
  }
  const float n = fmaxf(sqrtf(red[0]), 1e-12f);
  const float o = v / n;
  z[(size_t)i * FDIM + t]  = o;
  zh[(size_t)i * FDIM + t] = (_Float16)o;
}

// ---------------------------------------------------------------------------
// Exact diagonals: d11 = exp(2*z1.z1), d12 = exp(2*z1.z2), d22 = exp(2*z2.z2)
// ---------------------------------------------------------------------------
__global__ void diag_exp(const float* __restrict__ z1, const float* __restrict__ z2,
                         float* __restrict__ d11, float* __restrict__ d12,
                         float* __restrict__ d22) {
  __shared__ float r1[FDIM], r2[FDIM], r3[FDIM];
  const int i = blockIdx.x, t = threadIdx.x;
  const float a = z1[(size_t)i * FDIM + t];
  const float b = z2[(size_t)i * FDIM + t];
  r1[t] = a * a; r2[t] = a * b; r3[t] = b * b;
  __syncthreads();
  for (int s = FDIM / 2; s > 0; s >>= 1) {
    if (t < s) { r1[t] += r1[t + s]; r2[t] += r2[t + s]; r3[t] += r3[t + s]; }
    __syncthreads();
  }
  if (t == 0) {
    d11[i] = __expf(TEMP_INV * r1[0]);
    d12[i] = __expf(TEMP_INV * r2[0]);
    d22[i] = __expf(TEMP_INV * r3[0]);
  }
}

// ---------------------------------------------------------------------------
// Fused similarity sweep: rs[i] = sum_j exp(2 * dot(Za[i], Zb[j]))
// Block = 256 threads = 8 waves; each wave owns a 16-row strip (A cached in
// VGPRs).  All 8 waves share the B column tiles, so each 16x128 f16 tile is
// staged ONCE per block into LDS via CDNA5 async-to-LDS DMA
// (global_load_async_to_lds_b128, ASYNCcnt), double-buffered so tile ct+1
// streams in while tile ct feeds 4 WMMAs + fused exp/row-sum.  The N x N
// matrix is never materialized.  blockIdx.y selects the pass:
//   0:(z1,z1)->rs11  1:(z1,z2)->rs12  2:(z2,z1)->rs21  3:(z2,z2)->rs22
// ---------------------------------------------------------------------------
__global__ void simexp_rowsum(const _Float16* __restrict__ z1h,
                              const _Float16* __restrict__ z2h,
                              float* __restrict__ rs11, float* __restrict__ rs12,
                              float* __restrict__ rs21, float* __restrict__ rs22) {
  __shared__ __align__(16) _Float16 bt[2][16][BROW];   // 2 x 4.25KB buffers

  const int lane = threadIdx.x & 31;
  const int wave = threadIdx.x >> 5;
  const int tile = blockIdx.x * 8 + wave;              // 0..511 row strip
  const int g = lane >> 4;
  const int r = lane & 15;

  const _Float16* Ah; const _Float16* Bh; float* out;
  switch (blockIdx.y) {
    case 0:  Ah = z1h; Bh = z1h; out = rs11; break;
    case 1:  Ah = z1h; Bh = z2h; out = rs12; break;
    case 2:  Ah = z2h; Bh = z1h; out = rs21; break;
    default: Ah = z2h; Bh = z2h; out = rs22; break;
  }

  // Cache the full 16x128 A strip in registers (4 k-steps of v16h)
  const _Float16* arow = Ah + (size_t)(tile * 16 + r) * FDIM;
  v16h a[4];
#pragma unroll
  for (int ks = 0; ks < 4; ++ks) {
    v8h lo = *(const v8h*)(arow + ks * 32 + 8 * g);
    v8h hi = *(const v8h*)(arow + ks * 32 + 16 + 8 * g);
    a[ks] = __builtin_shufflevector(lo, hi, 0, 1, 2, 3, 4, 5, 6, 7,
                                    8, 9, 10, 11, 12, 13, 14, 15);
  }

  // Async stage of one 16x128 f16 tile: 256 threads x 16B = 4KB, one
  // global_load_async_to_lds_b128 per wave (8 per block per tile).
  const int srow = threadIdx.x >> 4;    // 0..15 tile row
  const int sseg = threadIdx.x & 15;    // 0..15 16B segment within row
  auto stage = [&](int buf, int ct) {
    unsigned ldsoff =
        (unsigned)(size_t)&bt[buf][srow][sseg * 8];   // low 32b = LDS byte addr
    unsigned long long src = (unsigned long long)(size_t)
        (Bh + (size_t)(ct * 16 + srow) * FDIM + sseg * 8);
    asm volatile("global_load_async_to_lds_b128 %0, %1, off"
                 :: "v"(ldsoff), "v"(src) : "memory");
  };

  float acc[8];
#pragma unroll
  for (int v = 0; v < 8; ++v) acc[v] = 0.0f;

  stage(0, 0);                                   // prime the pipeline
  for (int ct = 0; ct < NCT; ++ct) {
    const int buf = ct & 1;
    if (ct + 1 < NCT) {
      stage(buf ^ 1, ct + 1);                    // overlap next tile's DMA
      asm volatile("s_wait_asynccnt 0x1" ::: "memory");  // current tile done
    } else {
      asm volatile("s_wait_asynccnt 0x0" ::: "memory");
    }
    __syncthreads();                             // staged data visible to all

    v8f c = {};
#pragma unroll
    for (int ks = 0; ks < 4; ++ks) {
      v8h lo = *(const v8h*)&bt[buf][r][ks * 32 + 8 * g];
      v8h hi = *(const v8h*)&bt[buf][r][ks * 32 + 8 * g + 16];
      v16h b = __builtin_shufflevector(lo, hi, 0, 1, 2, 3, 4, 5, 6, 7,
                                       8, 9, 10, 11, 12, 13, 14, 15);
      c = __builtin_amdgcn_wmma_f32_16x16x32_f16(false, a[ks], false, b,
                                                 (short)0, c, false, false);
    }
#pragma unroll
    for (int v = 0; v < 8; ++v) acc[v] += __expf(TEMP_INV * c[v]);

    __syncthreads();                             // buffer reuse guard
  }

  // C tile layout: row = v + 8*g, col = r.  Reduce over the 16 lanes/group.
#pragma unroll
  for (int v = 0; v < 8; ++v) {
    float s = acc[v];
    for (int off = 8; off >= 1; off >>= 1) s += __shfl_xor(s, off, 16);
    if (r == 0) out[tile * 16 + 8 * g + v] = s;
  }
}

__global__ void zero_scalar(float* __restrict__ out) { out[0] = 0.0f; }

// ---------------------------------------------------------------------------
// loss_i = 0.5 * ( -log(d12/(rs11+rs12-d11)) - log(d12/(rs22+rs21-d22)) )
// mean over i, atomic accumulate into d_out[0]
// ---------------------------------------------------------------------------
__global__ void final_loss(const float* __restrict__ rs11, const float* __restrict__ rs12,
                           const float* __restrict__ rs21, const float* __restrict__ rs22,
                           const float* __restrict__ d11, const float* __restrict__ d12,
                           const float* __restrict__ d22, float* __restrict__ out) {
  __shared__ float red[256];
  const int i = blockIdx.x * blockDim.x + threadIdx.x;
  float v = 0.0f;
  if (i < NN) {
    const float l1 = -__logf(d12[i] / (rs11[i] + rs12[i] - d11[i]));
    const float l2 = -__logf(d12[i] / (rs22[i] + rs21[i] - d22[i]));
    v = 0.5f * (l1 + l2) * (1.0f / (float)NN);
  }
  red[threadIdx.x] = v;
  __syncthreads();
  for (int s = 128; s > 0; s >>= 1) {
    if (threadIdx.x < s) red[threadIdx.x] += red[threadIdx.x + s];
    __syncthreads();
  }
  if (threadIdx.x == 0) atomicAdd(out, red[0]);
}

// ---------------------------------------------------------------------------
// Host orchestration
// ---------------------------------------------------------------------------
static inline void launch_gemm(const float* A, const float* B, const float* bias,
                               float* C, int M, int K, int Nc, int act,
                               hipStream_t s) {
  const int tiles = (M / 16) * (Nc / 16);
  const int grid  = (tiles + 3) / 4;          // 4 waves / block (128 thr)
  gemm_wmma_f16<<<grid, 128, 0, s>>>(A, B, bias, C, M, K, Nc, act);
}

extern "C" void kernel_launch(void* const* d_in, const int* in_sizes, int n_in,
                              void* d_out, int out_size, void* d_ws, size_t ws_size,
                              hipStream_t stream) {
  (void)in_sizes; (void)n_in; (void)out_size; (void)ws_size;

  const float* feat1 = (const float*)d_in[0];
  const float* feat2 = (const float*)d_in[1];
  const int*   ei1   = (const int*)  d_in[2];
  const float* ew1   = (const float*)d_in[3];
  const int*   ei2   = (const int*)  d_in[4];
  const float* ew2   = (const float*)d_in[5];
  const float* w_l1a = (const float*)d_in[6];
  const float* b_l1a = (const float*)d_in[7];
  const float* w_l1b = (const float*)d_in[8];
  const float* b_l1b = (const float*)d_in[9];
  const float* w_g1  = (const float*)d_in[10];
  const float* b_g1  = (const float*)d_in[11];
  const float* w_g2  = (const float*)d_in[12];
  const float* b_g2  = (const float*)d_in[13];
  const float* w_fc1 = (const float*)d_in[14];
  const float* b_fc1 = (const float*)d_in[15];
  const float* w_fc2 = (const float*)d_in[16];
  const float* b_fc2 = (const float*)d_in[17];

  // Workspace carve-up (floats)
  float* ws   = (float*)d_ws;
  float* X1   = ws;              ws += (size_t)NN * 64;   // MLP hidden / proj hidden
  float* Xs   = ws;              ws += (size_t)NN * 32;   // encoder x
  float* SUP  = ws;              ws += (size_t)NN * 256;  // GCN support (max width)
  float* Hh   = ws;              ws += (size_t)NN * 256;  // GCN1 output
  float* Z1   = ws;              ws += (size_t)NN * 128;
  float* Z2   = ws;              ws += (size_t)NN * 128;
  float* ENC  = ws;              ws += (size_t)NN * 128;  // per-view encoder out
  _Float16* Z1H = (_Float16*)ws; ws += (size_t)NN * 64;   // f16 copies (2B each)
  _Float16* Z2H = (_Float16*)ws; ws += (size_t)NN * 64;
  float* RS11 = ws; ws += NN;
  float* RS12 = ws; ws += NN;
  float* RS21 = ws; ws += NN;
  float* RS22 = ws; ws += NN;
  float* D11  = ws; ws += NN;
  float* D12  = ws; ws += NN;
  float* D22  = ws; ws += NN;

  const int eb = (EE + 255) / 256;

  // ---- per-view encoder + projection ----
  for (int view = 0; view < 2; ++view) {
    const float* feat = view ? feat2 : feat1;
    const int*   ei   = view ? ei2   : ei1;
    const float* ew   = view ? ew2   : ew1;
    float*       Z    = view ? Z2    : Z1;

    // x = relu(feat @ w_l1a + b) @ w_l1b + b
    launch_gemm(feat, w_l1a, b_l1a, X1, NN, 512, 64, 1, stream);
    launch_gemm(X1, w_l1b, b_l1b, Xs, NN, 64, 32, 0, stream);

    // h = relu(gcn1(x))
    launch_gemm(Xs, w_g1, nullptr, SUP, NN, 32, 256, 0, stream);
    bias_init<<<(NN * 256 + 255) / 256, 256, 0, stream>>>(Hh, b_g1, 255, NN * 256);
    gcn_scatter<<<eb, 256, 0, stream>>>(ei, ew, SUP, Hh, EE, 256);
    relu_inplace<<<(NN * 256 + 255) / 256, 256, 0, stream>>>(Hh, NN * 256);

    // enc = gcn2(h)
    launch_gemm(Hh, w_g2, nullptr, SUP, NN, 256, 128, 0, stream);
    bias_init<<<(NN * 128 + 255) / 256, 256, 0, stream>>>(ENC, b_g2, 127, NN * 128);
    gcn_scatter<<<eb, 256, 0, stream>>>(ei, ew, SUP, ENC, EE, 128);

    // z = elu(enc @ w_fc1 + b) @ w_fc2 + b
    launch_gemm(ENC, w_fc1, b_fc1, X1, NN, 128, 64, 2, stream);
    launch_gemm(X1, w_fc2, b_fc2, Z, NN, 64, 128, 0, stream);
  }

  // ---- normalize + exact diagonals ----
  normalize_rows<<<NN, FDIM, 0, stream>>>(Z1, Z1H);
  normalize_rows<<<NN, FDIM, 0, stream>>>(Z2, Z2H);
  diag_exp<<<NN, FDIM, 0, stream>>>(Z1, Z2, D11, D12, D22);

  // ---- fused exp/row-sum similarity sweeps (4 passes via blockIdx.y) ----
  // 512 row strips per pass, 8 waves (256 thr) per block -> 64 blocks.x
  simexp_rowsum<<<dim3(64, 4), 256, 0, stream>>>(Z1H, Z2H, RS11, RS12, RS21, RS22);

  // ---- final scalar loss ----
  zero_scalar<<<1, 1, 0, stream>>>((float*)d_out);
  final_loss<<<NN / 256, 256, 0, stream>>>(RS11, RS12, RS21, RS22,
                                           D11, D12, D22, (float*)d_out);
}
